// SimpleParallelSequentialModel_21045339750572
// MI455X (gfx1250) — matrix operand order repaired and checked by reading, not compiled, not run on hardware
//
#include <hip/hip_runtime.h>

// Problem constants (from reference): B=4, L=8192, D=512
#define B_   4
#define L_   8192
#define D_   512
#define MTOT (B_ * L_)   // 32768 rows of the GEMM

typedef __attribute__((ext_vector_type(16))) __bf16 v16bf;
typedef __attribute__((ext_vector_type(8)))  float  v8f;

// ---------------------------------------------------------------------------
// Async global -> LDS copy (CDNA5): VDST = per-lane LDS byte address,
// VADDR = per-lane 32-bit byte offset (GVS mode), SADDR = 64-bit base.
// Tracked by ASYNCcnt; wave must s_wait_asynccnt before the barrier.
__device__ __forceinline__ void async_copy_b128(unsigned lds_off,
                                                unsigned long long base,
                                                unsigned goff)
{
    asm volatile("global_load_async_to_lds_b128 %0, %1, %2"
                 :: "v"(lds_off), "v"(goff), "s"(base) : "memory");
}
__device__ __forceinline__ void wait_async0()
{
    asm volatile("s_wait_asynccnt 0" ::: "memory");
}

// ---------------------------------------------------------------------------
// fp32 -> bf16 pre-conversion (one pass; removes all cvt from the GEMM loop)
__global__ __launch_bounds__(256)
void f32_to_bf16(const float* __restrict__ src, __bf16* __restrict__ dst)
{
    const size_t i = ((size_t)blockIdx.x * 256 + threadIdx.x) * 4;
    const float4 f = *(const float4*)(src + i);
    union { __bf16 h[4]; uint2 u; } t;
    t.h[0] = (__bf16)f.x; t.h[1] = (__bf16)f.y;
    t.h[2] = (__bf16)f.z; t.h[3] = (__bf16)f.w;
    *(uint2*)(dst + i) = t.u;
}

// ---------------------------------------------------------------------------
// GEMM: V[m, n] = sum_k X[m, k] * W[n, k]   (V = x @ W^T), bf16 WMMA, f32 acc
// Block tile 128(M) x 256(N), K-step 32. 256 threads = 8 waves (wave32),
// wave grid 2(M) x 4(N): each wave owns 64x64 = 4x4 tiles of 16x16 f32.
constexpr int BM = 128, BN = 256, BK = 32;
constexpr int LDT = BK + 8;          // padded LDS row stride (80 B, 16B-aligned)
constexpr int KT  = D_ / BK;         // 16 k-tiles

__global__ __launch_bounds__(256)
void gemm_xwt_bf16_wmma(const __bf16* __restrict__ Xb,
                        const __bf16* __restrict__ Wb,
                        float* __restrict__ V)
{
    __shared__ __align__(16) __bf16 lA[2][BM][LDT];
    __shared__ __align__(16) __bf16 lB[2][BN][LDT];

    const int tid   = threadIdx.x;
    const int lane  = tid & 31;
    const int wave  = tid >> 5;
    const int waveM = wave >> 2;      // 0..1 -> 64 rows each
    const int waveN = wave & 3;       // 0..3 -> 64 cols each
    const int bn    = blockIdx.x;     // 0..1   (N / 256)
    const int bm    = blockIdx.y;     // 0..255 (M / 128)

    const unsigned long long xbase = (unsigned long long)(uintptr_t)Xb;
    const unsigned long long wbase = (unsigned long long)(uintptr_t)Wb;

    // Async staging plan: 16-byte chunks (8 bf16). A tile = 512 chunks,
    // B tile = 1024 chunks; 256 threads -> 2 A-chunks + 4 B-chunks each.
    unsigned aLds[2][2], bLds[2][4];
    unsigned aG[2], bG[4];
#pragma unroll
    for (int j = 0; j < 2; ++j) {
        const int id = tid + 256 * j;          // 0..511
        const int r = id >> 2, q = id & 3;     // row 0..127, 16B chunk 0..3
        aG[j] = (unsigned)(((bm * BM + r) * D_ + q * 8) * 2);
#pragma unroll
        for (int buf = 0; buf < 2; ++buf)
            aLds[buf][j] = (unsigned)(uintptr_t)&lA[buf][r][q * 8];
    }
#pragma unroll
    for (int j = 0; j < 4; ++j) {
        const int id = tid + 256 * j;          // 0..1023
        const int r = id >> 2, q = id & 3;     // row 0..255
        bG[j] = (unsigned)(((bn * BN + r) * D_ + q * 8) * 2);
#pragma unroll
        for (int buf = 0; buf < 2; ++buf)
            bLds[buf][j] = (unsigned)(uintptr_t)&lB[buf][r][q * 8];
    }

    auto stage = [&](int buf, int k0) {
        const unsigned kb = (unsigned)(k0 * 2);
#pragma unroll
        for (int j = 0; j < 2; ++j) async_copy_b128(aLds[buf][j], xbase, aG[j] + kb);
#pragma unroll
        for (int j = 0; j < 4; ++j) async_copy_b128(bLds[buf][j], wbase, bG[j] + kb);
    };

    v8f acc[4][4] = {};

    // Fragment lane addressing (ISA 7.12.2):
    //  A 16x32 bf16: lanes 0-15 hold K=0-7 & 16-23; lanes 16-31: K=8-15 & 24-31
    //  B 32x16 bf16: lanes 0-15 hold K=0-7 & 16-23; lanes 16-31: K=8-15 & 24-31
    const int aRow = lane & 15;
    const int aK0  = (lane >> 4) * 8;
    const int bCol = lane & 15;
    const int bK0  = (lane >> 4) * 8;

    // prologue
    stage(0, 0);
    wait_async0();
    __syncthreads();

    for (int kt = 0; kt < KT; ++kt) {
        const int cur = kt & 1;
        const int nxt = cur ^ 1;

        if (kt + 1 < KT) stage(nxt, (kt + 1) * BK);   // DMA next tile

        v16bf afrag[4], bfrag[4];
#pragma unroll
        for (int mt = 0; mt < 4; ++mt) {
            const __bf16* p = &lA[cur][waveM * 64 + mt * 16 + aRow][aK0];
            union { uint4 u[2]; v16bf v; } t;
            t.u[0] = *(const uint4*)(p);        // K = aK0 .. aK0+7
            t.u[1] = *(const uint4*)(p + 16);   // K = aK0+16 .. aK0+23
            afrag[mt] = t.v;
        }
#pragma unroll
        for (int nt = 0; nt < 4; ++nt) {
            const __bf16* p = &lB[cur][waveN * 64 + nt * 16 + bCol][bK0];
            union { uint4 u[2]; v16bf v; } t;
            t.u[0] = *(const uint4*)(p);        // K = bK0 .. bK0+7
            t.u[1] = *(const uint4*)(p + 16);   // K = bK0+16 .. bK0+23
            bfrag[nt] = t.v;
        }
#pragma unroll
        for (int mt = 0; mt < 4; ++mt)
#pragma unroll
            for (int nt = 0; nt < 4; ++nt)
                acc[mt][nt] = __builtin_amdgcn_wmma_f32_16x16x32_bf16(
                    false, afrag[mt], false, bfrag[nt],
                    (short)0, acc[mt][nt], false, false);

        wait_async0();      // our async writes into `nxt` have landed in LDS
        __syncthreads();    // everyone's have; safe to read `nxt` next iter
    }

    // epilogue: C/D layout (7.12.2): lanes 0-15 -> M=r, lanes 16-31 -> M=8+r
    const int rbase = (lane >> 4) * 8;
    const int col   = lane & 15;
#pragma unroll
    for (int mt = 0; mt < 4; ++mt) {
        const int gm = bm * BM + waveM * 64 + mt * 16 + rbase;
#pragma unroll
        for (int nt = 0; nt < 4; ++nt) {
            const int gn = bn * BN + waveN * 64 + nt * 16 + col;
            float* o = V + (size_t)gm * D_ + gn;
#pragma unroll
            for (int r = 0; r < 8; ++r)
                o[(size_t)r * D_] = acc[mt][nt][r];
        }
    }
}

// ---------------------------------------------------------------------------
// Causal running mean: out = cumsum(V, L) / (l+1); chunked 3-pass scan.
constexpr int CH   = 64;
constexpr int CLEN = L_ / CH;        // 128

__global__ __launch_bounds__(128)
void chunk_sum(const float* __restrict__ V, float* __restrict__ S)
{
    const int dblk = blockIdx.x & 3;
    const int c    = (blockIdx.x >> 2) & (CH - 1);
    const int b    = blockIdx.x >> 8;
    const int d    = dblk * 128 + threadIdx.x;
    const float* p = V + ((size_t)(b * L_ + c * CLEN) * D_) + d;
    float s = 0.f;
#pragma unroll 4
    for (int i = 0; i < CLEN; ++i) s += p[(size_t)i * D_];
    S[((size_t)b * CH + c) * D_ + d] = s;
}

__global__ __launch_bounds__(256)
void chunk_scan(float* __restrict__ S)
{
    const int ch = blockIdx.x * 256 + threadIdx.x;   // 0..B*D-1
    const int b  = ch / D_;
    const int d  = ch % D_;
    float run = 0.f;
    for (int c = 0; c < CH; ++c) {
        const size_t idx = ((size_t)b * CH + c) * D_ + d;
        const float t = S[idx];
        S[idx] = run;        // exclusive prefix
        run += t;
    }
}

__global__ __launch_bounds__(128)
void chunk_apply(const float* __restrict__ V, const float* __restrict__ S,
                 float* __restrict__ out)
{
    const int dblk = blockIdx.x & 3;
    const int c    = (blockIdx.x >> 2) & (CH - 1);
    const int b    = blockIdx.x >> 8;
    const int d    = dblk * 128 + threadIdx.x;
    const size_t base = ((size_t)(b * L_ + c * CLEN) * D_) + d;
    float run = S[((size_t)b * CH + c) * D_ + d];
#pragma unroll 4
    for (int i = 0; i < CLEN; ++i) {
        run += V[base + (size_t)i * D_];
        const int l = c * CLEN + i;
        out[base + (size_t)i * D_] = run / (float)(l + 1);
    }
}

// ---------------------------------------------------------------------------
extern "C" void kernel_launch(void* const* d_in, const int* in_sizes, int n_in,
                              void* d_out, int out_size, void* d_ws, size_t ws_size,
                              hipStream_t stream)
{
    const float* X = (const float*)d_in[0];   // (B, L, D) fp32
    const float* W = (const float*)d_in[1];   // (D, D) fp32
    float* out = (float*)d_out;               // (B, L, D) fp32

    // workspace layout
    float*  V   = (float*)d_ws;                         // 64 MiB
    float*  S   = V + (size_t)B_ * L_ * D_;             // 512 KiB
    __bf16* Xb  = (__bf16*)(S + (size_t)B_ * CH * D_);  // 32 MiB
    __bf16* Wb  = Xb + (size_t)MTOT * D_;               // 512 KiB

    // 1) one-shot fp32 -> bf16 conversion of both GEMM operands
    f32_to_bf16<<<(MTOT * (size_t)D_) / (256 * 4), 256, 0, stream>>>(X, Xb);
    f32_to_bf16<<<((size_t)D_ * D_) / (256 * 4),   256, 0, stream>>>(W, Wb);

    // 2) WMMA GEMM with async global->LDS staging
    dim3 gg(D_ / BN, MTOT / BM);              // (2, 256)
    gemm_xwt_bf16_wmma<<<gg, 256, 0, stream>>>(Xb, Wb, V);

    // 3) causal running mean
    chunk_sum <<<B_ * CH * (D_ / 128), 128, 0, stream>>>(V, S);
    chunk_scan<<<(B_ * D_) / 256,      256, 0, stream>>>(S);
    chunk_apply<<<B_ * CH * (D_ / 128), 128, 0, stream>>>(V, S, out);
}
